// Embedding2Score_3135326126723
// MI455X (gfx1250) — matrix operand order repaired and checked by reading, not compiled, not run on hardware
//
#include <hip/hip_runtime.h>
#include <math.h>

typedef __attribute__((ext_vector_type(16))) __bf16 v16bf;
typedef __attribute__((ext_vector_type(8)))  float  v8f;

#define HDIM 128

// 16-bit A-matrix 16x32 layout (ISA 7.12.2): lane m = lane%16, hi = lane>>4
// element i (0..15) maps to k = i + (i&8) + 8*hi   (within the 32-wide K slab)
__device__ __forceinline__ int a_klocal(int i, int hi) { return i + (i & 8) + hi * 8; }

// ---------------------------------------------------------------------------
// Kernel 1: itemset embeddings. One wave (32 lanes) per itemset, lane owns 4
// contiguous floats of H=128. Masked gather-sum over P node rows, / len.
// ---------------------------------------------------------------------------
__global__ __launch_bounds__(256) void k_itemset_emb(
    const float* __restrict__ node_emb, const int* __restrict__ sequence,
    const int* __restrict__ itemset_len, float* __restrict__ itemset_emb,
    int T, int P, int NODESc, int ITEMSc) {
  const int wid  = threadIdx.x >> 5;
  const int lane = threadIdx.x & 31;
  const int t = blockIdx.x * 8 + wid;
  if (t >= T) return;
  const int s = t / ITEMSc;
  const size_t nbase = (size_t)s * NODESc;
  float4 acc = make_float4(0.f, 0.f, 0.f, 0.f);
  for (int p = 0; p < P; ++p) {
    const int sq = sequence[(size_t)t * P + p];
    if (sq < NODESc) {
      const float4 v = *(const float4*)(node_emb + (nbase + sq) * HDIM + lane * 4);
      acc.x += v.x; acc.y += v.y; acc.z += v.z; acc.w += v.w;
    }
  }
  const float inv = 1.0f / (float)itemset_len[t];
  acc.x *= inv; acc.y *= inv; acc.z *= inv; acc.w *= inv;
  *(float4*)(itemset_emb + (size_t)t * HDIM + lane * 4) = acc;
}

// ---------------------------------------------------------------------------
// Kernel 2: per-session attention head. 128 threads (4 waves) per session.
// gate[16,128] via 16x v_wmma_f32_16x16x32_bf16 per wave; alpha/s_g/s_h/y_hat
// with VALU + LDS reduction (work is negligible vs kernel 3).
// ---------------------------------------------------------------------------
__global__ __launch_bounds__(128) void k_session(
    const float* __restrict__ itemset_emb,
    const float* __restrict__ W1w, const float* __restrict__ W1b,
    const float* __restrict__ W2w, const float* __restrict__ W2b,
    const float* __restrict__ qw,  const float* __restrict__ qb,
    const float* __restrict__ W3w, const float* __restrict__ W3b,
    const float* __restrict__ embw, const int* __restrict__ cue,
    float* __restrict__ sh_out, float* __restrict__ yhat, int ITEMSc) {
  __shared__ float Xl[16 * HDIM];   // itemset embeddings of this session
  __shared__ float Gl[16 * HDIM];   // gate
  __shared__ float al[16];
  __shared__ float sgl[HDIM];
  __shared__ float shl[HDIM];
  __shared__ float red[HDIM];
  const int s = blockIdx.x;
  const int tid = threadIdx.x;
  const float* X = itemset_emb + (size_t)s * ITEMSc * HDIM;
  for (int i = tid; i < 16 * HDIM / 4; i += 128)
    ((float4*)Xl)[i] = ((const float4*)X)[i];
  __syncthreads();

  const int wid = tid >> 5, lane = tid & 31;
  const int m = lane & 15, hi = lane >> 4;
  const int vrow = ITEMSc - 1;  // v_n row

  // A fragments: X (per-itemset) and Vrep (broadcast last itemset)
  v16bf aX[4], aV[4];
#pragma unroll
  for (int ks = 0; ks < 4; ++ks) {
#pragma unroll
    for (int i = 0; i < 16; ++i) {
      const int k = ks * 32 + a_klocal(i, hi);
      aX[ks][i] = (__bf16)Xl[m * HDIM + k];
      aV[ks][i] = (__bf16)Xl[vrow * HDIM + k];
    }
  }

#pragma unroll
  for (int nt = 0; nt < 2; ++nt) {
    const int n0 = wid * 32 + nt * 16;
    const int n = n0 + m;  // output-feature row of W1/W2 handled by this lane pair
    v8f acc = {};
#pragma unroll
    for (int ks = 0; ks < 4; ++ks) {
      v16bf b1, b2;
#pragma unroll
      for (int i = 0; i < 16; ++i) {
        const int k = ks * 32 + a_klocal(i, hi);   // B layout: n=lane%16, k=i+(i&8)+8*hi
        b1[i] = (__bf16)W1w[(size_t)n * HDIM + k];
        b2[i] = (__bf16)W2w[(size_t)n * HDIM + k];
      }
      acc = __builtin_amdgcn_wmma_f32_16x16x32_bf16(false, aV[ks], false, b1,
                                                    (short)0, acc, false, false);
      acc = __builtin_amdgcn_wmma_f32_16x16x32_bf16(false, aX[ks], false, b2,
                                                    (short)0, acc, false, false);
    }
    const float bias = W1b[n] + W2b[n];
#pragma unroll
    for (int r = 0; r < 8; ++r) {
      const int mm = r + hi * 8;                    // D layout
      const float x = acc[r] + bias;
      Gl[mm * HDIM + n] = 1.0f / (1.0f + expf(-x));
    }
  }
  __syncthreads();

  if (tid < 16) {
    float a = qb[0];
    for (int h = 0; h < HDIM; ++h) a += Gl[tid * HDIM + h] * qw[h];
    al[tid] = a;
  }
  __syncthreads();
  {
    float v = 0.0f;
#pragma unroll
    for (int i = 0; i < 16; ++i) v += al[i] * Xl[i * HDIM + tid];
    sgl[tid] = v;
  }
  __syncthreads();
  {
    float v = W3b[tid];
    const float* wrow = W3w + (size_t)tid * 2 * HDIM;
    for (int j = 0; j < HDIM; ++j) v += wrow[j] * Xl[vrow * HDIM + j];
    for (int j = 0; j < HDIM; ++j) v += wrow[HDIM + j] * sgl[j];
    shl[tid] = v;
    sh_out[(size_t)s * HDIM + tid] = v;
  }
  __syncthreads();
  red[tid] = shl[tid] * embw[(size_t)cue[s] * HDIM + tid];
  __syncthreads();
  for (int off = 64; off > 0; off >>= 1) {
    if (tid < off) red[tid] += red[tid + off];
    __syncthreads();
  }
  if (tid == 0) yhat[s] = red[0];
}

// ---------------------------------------------------------------------------
// Kernel 3: all_scores = s_h @ emb_weight^T  ([B,128] x [V,128]^T).
// Write-bound (409.6 MB out vs 51.2 MB emb). Each block owns a disjoint
// 256-vocab-column slice (emb read once from HBM, stays L2-resident) x 256
// session rows. s_h panel staged in LDS as bf16 (64 KB); B-fragments held in
// VGPRs and reused across 16 M-tiles. Output streamed with non-temporal
// stores so the 409.6 MB write-once stream does not evict emb from L2.
// ---------------------------------------------------------------------------
__global__ __launch_bounds__(256) void k_scores(
    const float* __restrict__ sh, const float* __restrict__ embw,
    float* __restrict__ out, int Vn) {
  __shared__ __bf16 Abf[256 * HDIM];  // 64 KB
  const int tid = threadIdx.x;
  const int m0 = blockIdx.y * 256;
  for (int i = tid; i < 256 * HDIM / 4; i += 256) {
    const float4 f = ((const float4*)(sh + (size_t)m0 * HDIM))[i];
    Abf[i * 4 + 0] = (__bf16)f.x;
    Abf[i * 4 + 1] = (__bf16)f.y;
    Abf[i * 4 + 2] = (__bf16)f.z;
    Abf[i * 4 + 3] = (__bf16)f.w;
  }
  __syncthreads();

  const int wid = tid >> 5, lane = tid & 31;
  const int m = lane & 15, hi = lane >> 4;
  const int colbase = blockIdx.x * 256 + wid * 32;

  // Preload B fragments (2 n-subtiles x 4 k-steps); clamp+mask keeps all
  // lanes converged so EXEC is all-ones at every WMMA.
  v16bf b[2][4];
#pragma unroll
  for (int nt = 0; nt < 2; ++nt) {
    const int vr = colbase + nt * 16 + m;
    const float mask = (vr < Vn) ? 1.0f : 0.0f;
    const int vrc = (vr < Vn) ? vr : (Vn - 1);
    const float* src = embw + (size_t)vrc * HDIM + hi * 16;
#pragma unroll
    for (int ks = 0; ks < 4; ++ks) {
      const float4* s4 = (const float4*)(src + ks * 32);
      const float4 p0 = s4[0], p1 = s4[1], p2 = s4[2], p3 = s4[3];
      v16bf bb;
      bb[0]  = (__bf16)(p0.x * mask); bb[1]  = (__bf16)(p0.y * mask);
      bb[2]  = (__bf16)(p0.z * mask); bb[3]  = (__bf16)(p0.w * mask);
      bb[4]  = (__bf16)(p1.x * mask); bb[5]  = (__bf16)(p1.y * mask);
      bb[6]  = (__bf16)(p1.z * mask); bb[7]  = (__bf16)(p1.w * mask);
      bb[8]  = (__bf16)(p2.x * mask); bb[9]  = (__bf16)(p2.y * mask);
      bb[10] = (__bf16)(p2.z * mask); bb[11] = (__bf16)(p2.w * mask);
      bb[12] = (__bf16)(p3.x * mask); bb[13] = (__bf16)(p3.y * mask);
      bb[14] = (__bf16)(p3.z * mask); bb[15] = (__bf16)(p3.w * mask);
      b[nt][ks] = bb;
    }
  }

  // unroll-2 lets the scheduler overlap the next tile's ds_load_b128 A-frag
  // fetches with the current tile's WMMAs (B-frags are loop-invariant).
#pragma unroll 2
  for (int mt = 0; mt < 16; ++mt) {
    v16bf a[4];
#pragma unroll
    for (int ks = 0; ks < 4; ++ks)
#pragma unroll
      for (int i = 0; i < 16; ++i)
        a[ks][i] = Abf[(mt * 16 + m) * HDIM + ks * 32 + hi * 16 + i];
#pragma unroll
    for (int nt = 0; nt < 2; ++nt) {
      v8f acc = {};
#pragma unroll
      for (int ks = 0; ks < 4; ++ks)
        acc = __builtin_amdgcn_wmma_f32_16x16x32_bf16(false, a[ks], false, b[nt][ks],
                                                      (short)0, acc, false, false);
      const int col = colbase + nt * 16 + m;
      if (col < Vn) {
#pragma unroll
        for (int r = 0; r < 8; ++r) {
          const int row = m0 + mt * 16 + r + hi * 8;  // D layout
          __builtin_nontemporal_store(acc[r], &out[(size_t)row * Vn + col]);
        }
      }
    }
  }
}

// ---------------------------------------------------------------------------
extern "C" void kernel_launch(void* const* d_in, const int* in_sizes, int n_in,
                              void* d_out, int out_size, void* d_ws, size_t ws_size,
                              hipStream_t stream) {
  const float* node_emb = (const float*)d_in[0];
  const float* embw     = (const float*)d_in[1];
  const float* W1w      = (const float*)d_in[2];
  const float* W1b      = (const float*)d_in[3];
  const float* W2w      = (const float*)d_in[4];
  const float* W2b      = (const float*)d_in[5];
  const float* qw       = (const float*)d_in[6];
  const float* qb       = (const float*)d_in[7];
  const float* W3w      = (const float*)d_in[8];
  const float* W3b      = (const float*)d_in[9];
  const int* sequence    = (const int*)d_in[11];
  const int* itemset_len = (const int*)d_in[12];
  const int* cue         = (const int*)d_in[14];

  const int Bn     = in_sizes[13];          // 1024 sessions
  const int T      = in_sizes[12];          // 16384 itemsets
  const int Vn     = in_sizes[1] / HDIM;    // 100000 vocab
  const int Ntot   = in_sizes[0] / HDIM;    // 32768 nodes
  const int NODESc = Ntot / Bn;             // 32
  const int P      = in_sizes[11] / T;      // 8
  const int ITEMSc = T / Bn;                // 16

  float* itemset_emb = (float*)d_ws;                    // T*H floats
  float* s_h         = itemset_emb + (size_t)T * HDIM;  // B*H floats
  float* yhat   = (float*)d_out;
  float* scores = (float*)d_out + Bn;

  k_itemset_emb<<<(T + 7) / 8, 256, 0, stream>>>(node_emb, sequence, itemset_len,
                                                 itemset_emb, T, P, NODESc, ITEMSc);
  k_session<<<Bn, 128, 0, stream>>>(itemset_emb, W1w, W1b, W2w, W2b, qw, qb,
                                    W3w, W3b, embw, cue, s_h, yhat, ITEMSc);
  dim3 g3((Vn + 255) / 256, (Bn + 255) / 256);
  k_scores<<<g3, 256, 0, stream>>>(s_h, embw, scores, Vn);
}
